// net_76613626626502
// MI455X (gfx1250) — matrix operand order repaired and checked by reading, not compiled, not run on hardware
//
#include <hip/hip_runtime.h>

#define EPS 1e-6f

typedef __attribute__((ext_vector_type(16))) _Float16 v16h;
typedef __attribute__((ext_vector_type(8)))  _Float16 v8h;
typedef __attribute__((ext_vector_type(8)))  float    v8f;

static __device__ __forceinline__ _Float16 to_h(float x) {
  // clamp to f16 range: 1/x and exp features can overflow half
  x = fminf(fmaxf(x, -60000.0f), 60000.0f);
  return (_Float16)x;
}

// leaky(x) with slope 0.1 < 1:  max(x, 0.1*x)  (mul + max, no cmp/cndmask)
static __device__ __forceinline__ float leaky(float x) {
  return fmaxf(x, 0.1f * x);
}

// auge() 2-dim feature block layout (50 halves at o[0..49]):
// [data(2)@0, sin(17)@2, sqr(2)@19, cub(2)@21, exp(2)@23, inv(2)@25,
//  1/sin(17)@27, 1/sqr(2)@44, 1/cub(2)@46, 1/exp(2)@48]
static __device__ __forceinline__ void aug_sins(_Float16* o, float d0, float d1,
                                                const float* __restrict__ Ws,
                                                const float* __restrict__ bs,
                                                int lo, int hi) {
  for (int i = lo; i < hi; ++i) {
    float s = sinf(Ws[2 * i] * d0 + Ws[2 * i + 1] * d1 + bs[i]) + EPS;
    o[2 + i]  = to_h(s);
    o[27 + i] = to_h(1.0f / s);
  }
}

static __device__ __forceinline__ void aug_poly(_Float16* o, float d0, float d1) {
  float q0 = d0 * d0 + EPS, q1 = d1 * d1 + EPS;
  o[19] = to_h(q0); o[20] = to_h(q1); o[44] = to_h(1.0f / q0); o[45] = to_h(1.0f / q1);
  float c0 = d0 * d0 * d0 + EPS, c1 = d1 * d1 * d1 + EPS;
  o[21] = to_h(c0); o[22] = to_h(c1); o[46] = to_h(1.0f / c0); o[47] = to_h(1.0f / c1);
  float e0 = expf(d0) + EPS, e1 = expf(d1) + EPS;
  o[23] = to_h(e0); o[24] = to_h(e1); o[48] = to_h(1.0f / e0); o[49] = to_h(1.0f / e1);
  o[25] = to_h(1.0f / (d0 + EPS)); o[26] = to_h(1.0f / (d1 + EPS));
}

// -------- weight repack: W1[256,114], W2[128,256] (f32, row-major [out,in])
// --> f16 WMMA B fragments in ws.
// B-fragment layout (16-bit B 32x16, wave32): lane n holds column n%16;
// lane group g=lane/16 holds K=8g+h for h<8 and K=16+8g+(h-8) for h>=8.
// ws1: fragments f = nt*4 + kt   (nt<16, kt<4),  32768 halves at ws+0
// ws2: fragments f = nt*8 + kt   (nt<8,  kt<8),  32768 halves at ws+32768
__global__ __launch_bounds__(256) void prep_weights(const float* __restrict__ W1,
                                                    const float* __restrict__ W2,
                                                    _Float16* __restrict__ ws) {
  int tid = blockIdx.x * blockDim.x + threadIdx.x;   // 0..65535
  int which = tid >> 15;
  int r = tid & 32767;
  int h = r & 15;
  int lane = (r >> 4) & 31;
  int f = r >> 9;                                     // 0..63
  int group = lane >> 4;
  float val;
  if (which == 0) {
    int nt = f >> 2, kt = f & 3;
    int n = nt * 16 + (lane & 15);
    int k = kt * 32 + ((h & 8) << 1) + group * 8 + (h & 7);
    val = (k < 114) ? W1[n * 114 + k] : 0.0f;
  } else {
    int nt = f >> 3, kt = f & 7;
    int n = nt * 16 + (lane & 15);
    int k = kt * 32 + ((h & 8) << 1) + group * 8 + (h & 7);
    val = W2[n * 256 + k];
  }
  ws[tid] = (_Float16)val;
}

// -------- fused network: one workgroup (8 wave32) per batch element --------
__global__ __launch_bounds__(256) void fused_net(
    const float* __restrict__ ego_y, const float* __restrict__ Pn,
    const float* __restrict__ Pego,  const float* __restrict__ Vn,
    const float* __restrict__ Vego,  const float* __restrict__ Cn,
    const float* __restrict__ W_sel, const float* __restrict__ b_sel,
    const float* __restrict__ W_s0,  const float* __restrict__ b_s0,
    const float* __restrict__ W_map, const float* __restrict__ b_map,
    const float* __restrict__ W_efc, const float* __restrict__ b_efc,
    const float* __restrict__ W_s1,  const float* __restrict__ b_s1,
    const float* __restrict__ b1,    const float* __restrict__ b2,
    const float* __restrict__ W3,    const float* __restrict__ b3,
    const float* __restrict__ Ww,    const float* __restrict__ bw,
    const _Float16* __restrict__ ws, float* __restrict__ out) {
  // ldsA: GEMM1 A matrix [64 x 128] f16; reused as H2 [64 x 128] after GEMM1.
  __shared__ _Float16 ldsA[64 * 128];
  __shared__ _Float16 ldsH1[64 * 256];
  __shared__ float ldsEn[128];
  __shared__ float ldsLg[64];

  const int b    = blockIdx.x;
  const int tid  = threadIdx.x;
  const int lane = tid & 31;
  const int wave = tid >> 5;
  const int lm   = lane & 15;   // lane mod 16
  const int lg   = lane >> 4;   // lane group (0/1)

  // ------- feature rows: 4 threads per token, disjoint LDS writes ----------
  {
    const int t    = tid >> 2;      // token 0..63
    const int part = tid & 3;
    _Float16* row = &ldsA[t * 128];
    const int base = (b * 64 + t) * 2;
    if (part < 2) {
      float Pe0 = Pego[2 * b], Pe1 = Pego[2 * b + 1];
      float pn0 = Pn[base], pn1 = Pn[base + 1];
      float dp0 = pn0 - Pe0, dp1 = pn1 - Pe1;
      if (part == 0) {
        row[0] = to_h(Pe0); row[1] = to_h(Pe1);
        row[2] = to_h(pn0); row[3] = to_h(pn1);
        row[4] = to_h(dp0); row[5] = to_h(dp1);
        row[6] = to_h(dp0); row[7] = to_h(dp1);        // aug data copy
        aug_sins(row + 6, dp0, dp1, W_s1, b_s1, 0, 9);
      } else {
        aug_sins(row + 6, dp0, dp1, W_s1, b_s1, 9, 17);
        aug_poly(row + 6, dp0, dp1);
      }
    } else {
      float Ve0 = Vego[2 * b], Ve1 = Vego[2 * b + 1];
      float vn0 = Vn[base], vn1 = Vn[base + 1];
      float dv0 = vn0 - Ve0, dv1 = vn1 - Ve1;
      if (part == 2) {
        row[56] = to_h(Ve0); row[57] = to_h(Ve1);
        row[58] = to_h(vn0); row[59] = to_h(vn1);
        row[60] = to_h(dv0); row[61] = to_h(dv1);
        row[62] = to_h(dv0); row[63] = to_h(dv1);      // aug data copy
        aug_sins(row + 62, dv0, dv1, W_s1, b_s1, 0, 9);
      } else {
        aug_sins(row + 62, dv0, dv1, W_s1, b_s1, 9, 17);
        aug_poly(row + 62, dv0, dv1);
        row[112] = to_h(Cn[base]); row[113] = to_h(Cn[base + 1]);
        for (int k = 114; k < 128; ++k) row[k] = (_Float16)0.0f;
      }
    }
  }
  __syncthreads();

  // ---------------- GEMM1: [64,128] x [128,256] -> H1 (leaky) --------------
#pragma unroll 2
  for (int tl = wave; tl < 64; tl += 8) {
    int mt = tl >> 4, nt = tl & 15;
    v8f acc = {};
    const _Float16* Abase = &ldsA[(mt * 16 + lm) * 128 + 8 * lg];
    const _Float16* Bbase = &ws[((nt * 4) * 32 + lane) * 16];
#pragma unroll
    for (int kt = 0; kt < 4; ++kt) {
      v8h a0 = *(const v8h*)(Abase + kt * 32);
      v8h a1 = *(const v8h*)(Abase + kt * 32 + 16);
      v16h a = __builtin_shufflevector(a0, a1, 0, 1, 2, 3, 4, 5, 6, 7,
                                       8, 9, 10, 11, 12, 13, 14, 15);
      v16h bm = *(const v16h*)(Bbase + kt * 512);
      acc = __builtin_amdgcn_wmma_f32_16x16x32_f16(false, a, false, bm,
                                                   (short)0, acc, false, false);
    }
    float bias = b1[nt * 16 + lm];
#pragma unroll
    for (int r = 0; r < 8; ++r) {
      float v = leaky(acc[r] + bias);
      ldsH1[(mt * 16 + r + 8 * lg) * 256 + nt * 16 + lm] = to_h(v);
    }
  }
  __syncthreads();

  // ---------------- GEMM2: [64,256] x [256,128] -> H2 (leaky, in ldsA) -----
  const _Float16* ws2 = ws + 64 * 32 * 16;
#pragma unroll 2
  for (int tl = wave; tl < 32; tl += 8) {
    int mt = tl >> 3, nt = tl & 7;
    v8f acc = {};
    const _Float16* Abase = &ldsH1[(mt * 16 + lm) * 256 + 8 * lg];
    const _Float16* Bbase = &ws2[((nt * 8) * 32 + lane) * 16];
#pragma unroll
    for (int kt = 0; kt < 8; ++kt) {
      v8h a0 = *(const v8h*)(Abase + kt * 32);
      v8h a1 = *(const v8h*)(Abase + kt * 32 + 16);
      v16h a = __builtin_shufflevector(a0, a1, 0, 1, 2, 3, 4, 5, 6, 7,
                                       8, 9, 10, 11, 12, 13, 14, 15);
      v16h bm = *(const v16h*)(Bbase + kt * 512);
      acc = __builtin_amdgcn_wmma_f32_16x16x32_f16(false, a, false, bm,
                                                   (short)0, acc, false, false);
    }
    float bias = b2[nt * 16 + lm];
#pragma unroll
    for (int r = 0; r < 8; ++r) {
      float v = leaky(acc[r] + bias);
      ldsA[(mt * 16 + r + 8 * lg) * 128 + nt * 16 + lm] = to_h(v);
    }
  }
  __syncthreads();

  // ---------------- GEMM3 (K=128,N=2) + logits (threads 0..63) -------------
  if (tid < 64) {
    const int t = tid;
    float s0 = b3[0], s1 = b3[1];
    const _Float16* hrow = &ldsA[t * 128];
#pragma unroll 4
    for (int k = 0; k < 128; ++k) {
      float hv = (float)hrow[k];
      s0 += hv * W3[k];
      s1 += hv * W3[128 + k];
    }
    int base = (b * 64 + t) * 2;
    float pn0 = Pn[base], pn1 = Pn[base + 1];
    float dp0 = pn0 - Pego[2 * b], dp1 = pn1 - Pego[2 * b + 1];
    float lgt = Ww[0] * s0 + Ww[1] * s1 + Ww[2] * pn0 + Ww[3] * pn1 +
                Ww[4] * dp0 + Ww[5] * dp1 + bw[0];
    ldsEn[2 * t] = s0; ldsEn[2 * t + 1] = s1; ldsLg[t] = lgt;
  }
  __syncthreads();

  // ------- softmax over 64 neighbors: wave-0 parallel reduction ------------
  if (tid < 32) {
    const int t = tid;
    float l0 = ldsLg[t], l1 = ldsLg[t + 32];
    float m = fmaxf(l0, l1);
#pragma unroll
    for (int off = 16; off > 0; off >>= 1) m = fmaxf(m, __shfl_xor(m, off, 32));
    float e0 = expf(l0 - m), e1 = expf(l1 - m);
    float s  = e0 + e1;
    float a0 = e0 * ldsEn[2 * t]     + e1 * ldsEn[2 * (t + 32)];
    float a1 = e0 * ldsEn[2 * t + 1] + e1 * ldsEn[2 * (t + 32) + 1];
#pragma unroll
    for (int off = 16; off > 0; off >>= 1) {
      s  += __shfl_xor(s, off, 32);
      a0 += __shfl_xor(a0, off, 32);
      a1 += __shfl_xor(a1, off, 32);
    }
    if (tid == 0) {
      out[4 * b + 2] = a0 / s;
      out[4 * b + 3] = a1 / s;
    }
  }

  // ---------------- Er net (tiny, thread 32 / wave 1, f32 VALU) ------------
  if (tid == 32) {
    float ey = ego_y[b];
    float dy[4] = {13.55f - ey, 17.45f - ey, 21.12f - ey, 24.91f - ey};
    float z[4];
#pragma unroll
    for (int o = 0; o < 4; ++o) {
      float s = b_sel[o];
#pragma unroll
      for (int j = 0; j < 4; ++j) s += W_sel[o * 4 + j] * dy[j];
      z[o] = s;
    }
    float zm = fmaxf(fmaxf(z[0], z[1]), fmaxf(z[2], z[3]));
    float es[4], se = 0.0f;
#pragma unroll
    for (int o = 0; o < 4; ++o) { es[o] = expf(z[o] - zm); se += es[o]; }
    float x[4];
#pragma unroll
    for (int o = 0; o < 4; ++o) x[o] = (es[o] / se) * dy[o];
    // auge(x) -> 48 features
    float g[48];
#pragma unroll
    for (int j = 0; j < 4; ++j) g[j] = x[j];
#pragma unroll
    for (int i = 0; i < 8; ++i) {
      float s = b_s0[i];
#pragma unroll
      for (int j = 0; j < 4; ++j) s += W_s0[i * 4 + j] * x[j];
      s = sinf(s) + EPS;
      g[4 + i] = s; g[28 + i] = 1.0f / s;
    }
#pragma unroll
    for (int j = 0; j < 4; ++j) {
      float q = x[j] * x[j] + EPS;          g[12 + j] = q; g[36 + j] = 1.0f / q;
      float c = x[j] * x[j] * x[j] + EPS;   g[16 + j] = c; g[40 + j] = 1.0f / c;
      float e = expf(x[j]) + EPS;           g[20 + j] = e; g[44 + j] = 1.0f / e;
      g[24 + j] = 1.0f / (x[j] + EPS);
    }
    float er0 = b_efc[0], er1 = b_efc[1];
#pragma unroll
    for (int o = 0; o < 8; ++o) {
      float m = b_map[o];
      for (int k = 0; k < 48; ++k) m += W_map[o * 48 + k] * g[k];
      m = leaky(m);
      er0 += W_efc[o] * m;
      er1 += W_efc[8 + o] * m;
    }
    out[4 * b + 0] = er0;
    out[4 * b + 1] = er1;
  }
}

extern "C" void kernel_launch(void* const* d_in, const int* in_sizes, int n_in,
                              void* d_out, int out_size, void* d_ws, size_t ws_size,
                              hipStream_t stream) {
  (void)in_sizes; (void)n_in; (void)out_size; (void)ws_size;
  const float* ego_y = (const float*)d_in[0];
  const float* Pn    = (const float*)d_in[1];
  const float* Pego  = (const float*)d_in[2];
  const float* Vn    = (const float*)d_in[3];
  const float* Vego  = (const float*)d_in[4];
  const float* Cn    = (const float*)d_in[5];
  const float* W_sel = (const float*)d_in[6];
  const float* b_sel = (const float*)d_in[7];
  const float* W_s0  = (const float*)d_in[8];
  const float* b_s0  = (const float*)d_in[9];
  const float* W_map = (const float*)d_in[10];
  const float* b_map = (const float*)d_in[11];
  const float* W_efc = (const float*)d_in[12];
  const float* b_efc = (const float*)d_in[13];
  const float* W_s1  = (const float*)d_in[14];
  const float* b_s1  = (const float*)d_in[15];
  const float* W1    = (const float*)d_in[16];
  const float* b1    = (const float*)d_in[17];
  const float* W2    = (const float*)d_in[18];
  const float* b2    = (const float*)d_in[19];
  const float* W3    = (const float*)d_in[20];
  const float* b3    = (const float*)d_in[21];
  const float* Ww    = (const float*)d_in[22];
  const float* bw    = (const float*)d_in[23];
  float* out = (float*)d_out;
  _Float16* ws = (_Float16*)d_ws;

  // repack weights into WMMA B-fragment layout (128 KB in d_ws)
  prep_weights<<<256, 256, 0, stream>>>(W1, W2, ws);
  // fused network: one workgroup per batch element
  fused_net<<<4096, 256, 0, stream>>>(ego_y, Pn, Pego, Vn, Vego, Cn,
                                      W_sel, b_sel, W_s0, b_s0, W_map, b_map,
                                      W_efc, b_efc, W_s1, b_s1,
                                      b1, b2, W3, b3, Ww, bw, ws, out);
}
